// MOE_simple_30425548325141
// MI455X (gfx1250) — matrix operand order repaired and checked by reading, not compiled, not run on hardware
//
#include <hip/hip_runtime.h>
#include <hip/hip_bf16.h>

// Problem constants (match reference)
#define B_TOK 8192
#define C_DIM 1024
#define E_NUM 8
#define H_DIM 4096
#define TOPK_K 4
#define LN_EPS 1e-6f

typedef __bf16 bf16;
typedef __attribute__((ext_vector_type(8)))  bf16  bf16x8;
typedef __attribute__((ext_vector_type(16))) bf16  bf16x16;
typedef __attribute__((ext_vector_type(8)))  float v8f;
typedef int v4i __attribute__((vector_size(16)));   // GCC-style vector: matches builtin decl

union ABfrag { bf16x16 v; bf16x8 h[2]; };

// ---------------------------------------------------------------------------
// CDNA5 async global->LDS copy path (guarded so either toolchain compiles)
// Builtin signature (from hipcc diagnostic):
//   void __builtin_amdgcn_global_load_async_to_lds_b128(
//       __attribute__((address_space(1))) v4i* gsrc,   // printed as "__device__"
//       __attribute__((address_space(3))) v4i* ldst,   // printed as "__shared__"
//       imm int offset, imm int cpol)
// ---------------------------------------------------------------------------
#if defined(__has_builtin)
  #if __has_builtin(__builtin_amdgcn_global_load_async_to_lds_b128)
    #define USE_ASYNC_COPY 1
  #endif
  #if __has_builtin(__builtin_amdgcn_s_wait_asynccnt)
    #define HAVE_WAIT_ASYNC_BUILTIN 1
  #endif
#endif

#define AS1 __attribute__((address_space(1)))
#define AS3 __attribute__((address_space(3)))

__device__ __forceinline__ AS3 v4i* lds_cast_v4(void* p) {
    // generic LDS pointer: low 32 bits are the LDS byte offset (ISA 10.2)
    return (AS3 v4i*)(unsigned int)(unsigned long long)p;
}
__device__ __forceinline__ AS1 v4i* gbl_cast_v4(const void* p) {
    // global flat address == addrspace(1) address (identity mapping)
    return (AS1 v4i*)(unsigned long long)p;
}

__device__ __forceinline__ void cp_b128(void* lds_dst, const void* gsrc) {
#ifdef USE_ASYNC_COPY
    __builtin_amdgcn_global_load_async_to_lds_b128(gbl_cast_v4(gsrc), lds_cast_v4(lds_dst), 0, 0);
#else
    *(bf16x8*)lds_dst = *(const bf16x8*)gsrc;   // sync fallback: vmem load + ds_store_b128
#endif
}

__device__ __forceinline__ void wait_stage() {
#ifdef USE_ASYNC_COPY
  #ifdef HAVE_WAIT_ASYNC_BUILTIN
    __builtin_amdgcn_s_wait_asynccnt(0);
  #else
    asm volatile("s_wait_asynccnt 0x0" ::: "memory");
  #endif
#endif
}

__device__ __forceinline__ float quick_gelu(float v) {
    return v * (1.0f / (1.0f + __expf(-1.702f * v)));
}

// ---------------------------------------------------------------------------
// fp32 -> bf16 conversions (+ transposes so WMMA B-fragments are contiguous)
// ---------------------------------------------------------------------------
__global__ void cvt_x_kernel(const float* __restrict__ in, bf16* __restrict__ out, int n) {
    for (int i = blockIdx.x * blockDim.x + threadIdx.x; i < n; i += gridDim.x * blockDim.x)
        out[i] = (bf16)in[i];
}

// W1 [E][C][H] fp32 -> w1t [E][H][C] bf16
__global__ void cvt_w1_kernel(const float* __restrict__ in, bf16* __restrict__ out) {
    const int n = E_NUM * C_DIM * H_DIM;
    for (int i = blockIdx.x * blockDim.x + threadIdx.x; i < n; i += gridDim.x * blockDim.x) {
        int e = i / (C_DIM * H_DIM);
        int r = i - e * (C_DIM * H_DIM);
        int c = r / H_DIM;
        int h = r - c * H_DIM;
        out[(size_t)e * H_DIM * C_DIM + (size_t)h * C_DIM + c] = (bf16)in[i];
    }
}

// W2 [E][H][C] fp32 -> w2t [E][C][H] bf16
__global__ void cvt_w2_kernel(const float* __restrict__ in, bf16* __restrict__ out) {
    const int n = E_NUM * H_DIM * C_DIM;
    for (int i = blockIdx.x * blockDim.x + threadIdx.x; i < n; i += gridDim.x * blockDim.x) {
        int e = i / (H_DIM * C_DIM);
        int r = i - e * (H_DIM * C_DIM);
        int h = r / C_DIM;
        int c = r - h * C_DIM;
        out[(size_t)e * C_DIM * H_DIM + (size_t)c * H_DIM + h] = (bf16)in[i];
    }
}

// ---------------------------------------------------------------------------
// Gate: softmax(x@Wg + bg) -> top-4 mask -> dense per-token weights [B][E]
// One wave32 per token.
// ---------------------------------------------------------------------------
__global__ __launch_bounds__(256) void gate_kernel(const float* __restrict__ x,
                                                   const float* __restrict__ Wg,
                                                   const float* __restrict__ bg,
                                                   float* __restrict__ gw) {
    const int wid  = threadIdx.x >> 5;
    const int lane = threadIdx.x & 31;
    const int b    = blockIdx.x * 8 + wid;
    if (b >= B_TOK) return;

    float acc[E_NUM] = {0.f, 0.f, 0.f, 0.f, 0.f, 0.f, 0.f, 0.f};
    const float* xr = x + (size_t)b * C_DIM;
    for (int j = 0; j < C_DIM; j += 32) {
        int c = j + lane;
        float xv = xr[c];
        const float4 w0 = *(const float4*)(Wg + (size_t)c * E_NUM);
        const float4 w1 = *(const float4*)(Wg + (size_t)c * E_NUM + 4);
        acc[0] += xv * w0.x; acc[1] += xv * w0.y; acc[2] += xv * w0.z; acc[3] += xv * w0.w;
        acc[4] += xv * w1.x; acc[5] += xv * w1.y; acc[6] += xv * w1.z; acc[7] += xv * w1.w;
    }
    #pragma unroll
    for (int off = 16; off > 0; off >>= 1) {
        #pragma unroll
        for (int e = 0; e < E_NUM; ++e)
            acc[e] += __shfl_xor(acc[e], off, 32);
    }
    if (lane == 0) {
        float s[E_NUM];
        float mx = -3.4e38f;
        #pragma unroll
        for (int e = 0; e < E_NUM; ++e) { s[e] = acc[e] + bg[e]; mx = fmaxf(mx, s[e]); }
        float sum = 0.f;
        #pragma unroll
        for (int e = 0; e < E_NUM; ++e) { s[e] = __expf(s[e] - mx); sum += s[e]; }
        float inv = 1.0f / sum;
        #pragma unroll
        for (int e = 0; e < E_NUM; ++e) s[e] *= inv;
        float w[E_NUM] = {0.f, 0.f, 0.f, 0.f, 0.f, 0.f, 0.f, 0.f};
        float t[E_NUM];
        #pragma unroll
        for (int e = 0; e < E_NUM; ++e) t[e] = s[e];
        for (int k = 0; k < TOPK_K; ++k) {
            int am = 0; float av = t[0];
            #pragma unroll
            for (int e = 1; e < E_NUM; ++e) if (t[e] > av) { av = t[e]; am = e; }
            w[am] = s[am];
            t[am] = -1.0f;
        }
        #pragma unroll
        for (int e = 0; e < E_NUM; ++e) gw[(size_t)b * E_NUM + e] = w[e];
    }
}

// ---------------------------------------------------------------------------
// Cooperative block GEMM machinery:
//   block = 256 threads = 8 waves in a 2(M) x 4(N) grid
//   block tile = 64(M) x 256(N); wave tile = 32x64 = 2x4 WMMA 16x16 tiles
//   per K-step (32): stage A(64x32) + B(256x32) bf16 into double-buffered LDS
// ---------------------------------------------------------------------------
__device__ __forceinline__ void stage_tiles(const bf16* __restrict__ A0,
                                            const bf16* __restrict__ B0,
                                            int ldA, int ldB, int k,
                                            bf16* As, bf16* Bs, int tid) {
    // A tile: 64 rows x 32 K. 256 threads -> each copies 16B.
    {
        const int row = tid >> 2;
        const int cg  = (tid & 3) * 8;
        cp_b128(&As[row * 32 + cg], A0 + (size_t)row * ldA + k + cg);
    }
    // B tile: 256 cols x 32 K (col-major-by-n). Each thread copies one col = 64B.
    {
        const int n = tid;
        const bf16* src = B0 + (size_t)n * ldB + k;
        bf16* dst = &Bs[n * 32];
        #pragma unroll
        for (int j = 0; j < 4; ++j)
            cp_b128(dst + j * 8, src + j * 8);
        __builtin_prefetch((const void*)(src + 64), 0, 1);   // k+64 stream hint
    }
}

__device__ __forceinline__ void compute_step(const bf16* As, const bf16* Bs,
                                             int wm, int wn, int r, int hi,
                                             v8f acc[2][4]) {
    ABfrag a[2];
    #pragma unroll
    for (int mi = 0; mi < 2; ++mi) {
        const bf16* p = As + (size_t)(wm * 32 + mi * 16 + r) * 32;
        a[mi].h[0] = *(const bf16x8*)(p + hi * 16);       // K {0..7}/{16..23}
        a[mi].h[1] = *(const bf16x8*)(p + hi * 16 + 8);   // K {8..15}/{24..31}
    }
    #pragma unroll
    for (int ni = 0; ni < 4; ++ni) {
        const bf16* p = Bs + (size_t)(wn * 64 + ni * 16 + r) * 32 + hi * 16;
        bf16x16 bv = *(const bf16x16*)p;                  // 16 contiguous K values
        #pragma unroll
        for (int mi = 0; mi < 2; ++mi)
            acc[mi][ni] = __builtin_amdgcn_wmma_f32_16x16x32_bf16(
                false, a[mi].v, false, bv, (short)0, acc[mi][ni], false, false);
    }
}

// ---------------------------------------------------------------------------
// MLP layer 1: h[e,b,:] = quickGELU(x @ W1[e] + b1[e]), bf16 out [E][B][H]
// ---------------------------------------------------------------------------
__global__ __launch_bounds__(256) void mlp1_kernel(const bf16* __restrict__ xb,
                                                   const bf16* __restrict__ w1t,
                                                   const float* __restrict__ b1,
                                                   bf16* __restrict__ hb) {
    __shared__ __align__(16) bf16 As[2][64 * 32];
    __shared__ __align__(16) bf16 Bs[2][256 * 32];

    const int tid  = threadIdx.x;
    const int wid  = tid >> 5;
    const int lane = tid & 31;
    const int wm = wid >> 2, wn = wid & 3;
    const int r  = lane & 15, hi = lane >> 4;

    const int TMB = B_TOK / 64;    // 128
    const int TNB = H_DIM / 256;   // 16
    const int e  = blockIdx.x / (TMB * TNB);
    const int t  = blockIdx.x % (TMB * TNB);
    const int m0 = (t / TNB) * 64;
    const int n0 = (t % TNB) * 256;

    const bf16* A0 = xb  + (size_t)m0 * C_DIM;
    const bf16* B0 = w1t + (size_t)e * H_DIM * C_DIM + (size_t)n0 * C_DIM;

    v8f acc[2][4] = {};

    int cur = 0;
    stage_tiles(A0, B0, C_DIM, C_DIM, 0, As[0], Bs[0], tid);
    wait_stage();
    __syncthreads();
    for (int k = 0; k < C_DIM; k += 32) {
        if (k + 32 < C_DIM)
            stage_tiles(A0, B0, C_DIM, C_DIM, k + 32, As[cur ^ 1], Bs[cur ^ 1], tid);
        compute_step(As[cur], Bs[cur], wm, wn, r, hi, acc);
        wait_stage();
        __syncthreads();
        cur ^= 1;
    }

    // Epilogue: bias + quickGELU, store bf16 h[e][m][n]
    #pragma unroll
    for (int mi = 0; mi < 2; ++mi) {
        #pragma unroll
        for (int ni = 0; ni < 4; ++ni) {
            const int n = n0 + wn * 64 + ni * 16 + r;
            const float bias = b1[(size_t)e * H_DIM + n];
            #pragma unroll
            for (int i = 0; i < 8; ++i) {
                const int m = m0 + wm * 32 + mi * 16 + i + hi * 8;  // C/D VGPR layout
                float v = acc[mi][ni][i] + bias;
                hb[((size_t)e * B_TOK + m) * H_DIM + n] = (bf16)quick_gelu(v);
            }
        }
    }
}

// ---------------------------------------------------------------------------
// MLP layer 2: y[b,e,:] = quickGELU(h[e,b,:] @ W2[e] + b2[e]), fp32 [B][E][C]
// ---------------------------------------------------------------------------
__global__ __launch_bounds__(256) void mlp2_kernel(const bf16* __restrict__ hb,
                                                   const bf16* __restrict__ w2t,
                                                   const float* __restrict__ b2,
                                                   float* __restrict__ yb) {
    __shared__ __align__(16) bf16 As[2][64 * 32];
    __shared__ __align__(16) bf16 Bs[2][256 * 32];

    const int tid  = threadIdx.x;
    const int wid  = tid >> 5;
    const int lane = tid & 31;
    const int wm = wid >> 2, wn = wid & 3;
    const int r  = lane & 15, hi = lane >> 4;

    const int TMB = B_TOK / 64;    // 128
    const int TNB = C_DIM / 256;   // 4
    const int e  = blockIdx.x / (TMB * TNB);
    const int t  = blockIdx.x % (TMB * TNB);
    const int m0 = (t / TNB) * 64;
    const int n0 = (t % TNB) * 256;

    const bf16* A0 = hb  + ((size_t)e * B_TOK + m0) * H_DIM;
    const bf16* B0 = w2t + (size_t)e * C_DIM * H_DIM + (size_t)n0 * H_DIM;

    v8f acc[2][4] = {};

    int cur = 0;
    stage_tiles(A0, B0, H_DIM, H_DIM, 0, As[0], Bs[0], tid);
    wait_stage();
    __syncthreads();
    for (int k = 0; k < H_DIM; k += 32) {
        if (k + 32 < H_DIM)
            stage_tiles(A0, B0, H_DIM, H_DIM, k + 32, As[cur ^ 1], Bs[cur ^ 1], tid);
        compute_step(As[cur], Bs[cur], wm, wn, r, hi, acc);
        wait_stage();
        __syncthreads();
        cur ^= 1;
    }

    #pragma unroll
    for (int mi = 0; mi < 2; ++mi) {
        #pragma unroll
        for (int ni = 0; ni < 4; ++ni) {
            const int n = n0 + wn * 64 + ni * 16 + r;
            const float bias = b2[(size_t)e * C_DIM + n];
            #pragma unroll
            for (int i = 0; i < 8; ++i) {
                const int m = m0 + wm * 32 + mi * 16 + i + hi * 8;
                float v = acc[mi][ni][i] + bias;
                yb[((size_t)m * E_NUM + e) * C_DIM + n] = quick_gelu(v);
            }
        }
    }
}

// ---------------------------------------------------------------------------
// Combine: per token, layernorm each selected expert row and weighted-sum.
// ---------------------------------------------------------------------------
__global__ __launch_bounds__(256) void combine_kernel(const float* __restrict__ yb,
                                                      const float* __restrict__ gw,
                                                      const float* __restrict__ ln_g,
                                                      const float* __restrict__ ln_b,
                                                      float* __restrict__ out) {
    const int b    = blockIdx.x;
    const int tid  = threadIdx.x;
    const int wid  = tid >> 5;
    const int lane = tid & 31;
    const int c0   = tid * 4;

    __shared__ float2 s_red[8];

    float acc[4] = {0.f, 0.f, 0.f, 0.f};

    for (int e = 0; e < E_NUM; ++e) {
        const float w = gw[(size_t)b * E_NUM + e];   // uniform across block
        if (w == 0.0f) continue;                     // uniform branch: skip unselected
        const float* row = yb + ((size_t)b * E_NUM + e) * C_DIM;
        const float4 v = *(const float4*)(row + c0);
        float s = v.x + v.y + v.z + v.w;
        float q = v.x * v.x + v.y * v.y + v.z * v.z + v.w * v.w;
        #pragma unroll
        for (int off = 16; off > 0; off >>= 1) {
            s += __shfl_xor(s, off, 32);
            q += __shfl_xor(q, off, 32);
        }
        if (lane == 0) s_red[wid] = make_float2(s, q);
        __syncthreads();
        float ts = 0.f, tq = 0.f;
        #pragma unroll
        for (int i = 0; i < 8; ++i) { ts += s_red[i].x; tq += s_red[i].y; }
        __syncthreads();
        const float mu   = ts * (1.0f / C_DIM);
        const float var  = tq * (1.0f / C_DIM) - mu * mu;
        const float rinv = rsqrtf(var + LN_EPS);
        const float* g  = ln_g + (size_t)e * C_DIM + c0;
        const float* bb = ln_b + (size_t)e * C_DIM + c0;
        const float vv[4] = {v.x, v.y, v.z, v.w};
        #pragma unroll
        for (int j = 0; j < 4; ++j)
            acc[j] += w * ((vv[j] - mu) * rinv * g[j] + bb[j]);
    }
    float4 o = make_float4(acc[0], acc[1], acc[2], acc[3]);
    *(float4*)(out + (size_t)b * C_DIM + c0) = o;
}

// ---------------------------------------------------------------------------
extern "C" void kernel_launch(void* const* d_in, const int* in_sizes, int n_in,
                              void* d_out, int out_size, void* d_ws, size_t ws_size,
                              hipStream_t stream) {
    const float* x    = (const float*)d_in[0];
    const float* W1   = (const float*)d_in[1];
    const float* b1   = (const float*)d_in[2];
    const float* W2   = (const float*)d_in[3];
    const float* b2   = (const float*)d_in[4];
    const float* ln_g = (const float*)d_in[5];
    const float* ln_b = (const float*)d_in[6];
    const float* Wg   = (const float*)d_in[7];
    const float* bg   = (const float*)d_in[8];
    float* out = (float*)d_out;

    // Workspace layout (all sizes multiples of 256B)
    char* ws = (char*)d_ws;
    size_t o = 0;
    bf16*  xb  = (bf16*)(ws + o);  o += (size_t)B_TOK * C_DIM * sizeof(bf16);           //  16 MB
    bf16*  w1t = (bf16*)(ws + o);  o += (size_t)E_NUM * H_DIM * C_DIM * sizeof(bf16);   //  64 MB
    bf16*  w2t = (bf16*)(ws + o);  o += (size_t)E_NUM * C_DIM * H_DIM * sizeof(bf16);   //  64 MB
    bf16*  hb  = (bf16*)(ws + o);  o += (size_t)E_NUM * B_TOK * H_DIM * sizeof(bf16);   // 512 MB
    float* yb  = (float*)(ws + o); o += (size_t)B_TOK * E_NUM * C_DIM * sizeof(float);  // 256 MB
    float* gw  = (float*)(ws + o); o += (size_t)B_TOK * E_NUM * sizeof(float);          // 256 KB

    // 1) convert / transpose to bf16
    cvt_x_kernel<<<4096, 256, 0, stream>>>(x, xb, B_TOK * C_DIM);
    cvt_w1_kernel<<<8192, 256, 0, stream>>>(W1, w1t);
    cvt_w2_kernel<<<8192, 256, 0, stream>>>(W2, w2t);

    // 2) gate (one wave per token)
    gate_kernel<<<B_TOK / 8, 256, 0, stream>>>(x, Wg, bg, gw);

    // 3) expert MLP layer 1: E * (B/64) * (H/256) = 16384 blocks
    mlp1_kernel<<<E_NUM * (B_TOK / 64) * (H_DIM / 256), 256, 0, stream>>>(xb, w1t, b1, hb);

    // 4) expert MLP layer 2: E * (B/64) * (C/256) = 4096 blocks
    mlp2_kernel<<<E_NUM * (B_TOK / 64) * (C_DIM / 256), 256, 0, stream>>>(hb, w2t, b2, yb);

    // 5) layernorm + top-k weighted combine
    combine_kernel<<<B_TOK, 256, 0, stream>>>(yb, gw, ln_g, ln_b, out);
}
